// EncoderBlock_25434796327553
// MI455X (gfx1250) — matrix operand order repaired and checked by reading, not compiled, not run on hardware
//
#include <hip/hip_runtime.h>
#include <math.h>

// ---------------------------------------------------------------------------
// CDNA5 (gfx1250) transformer encoder block:
//   bf16 WMMA GEMMs + flash attention, tiles staged via Tensor Data Mover
// ---------------------------------------------------------------------------

typedef __bf16 bf16;
typedef __attribute__((ext_vector_type(16))) __bf16 v16bf;
typedef __attribute__((ext_vector_type(8)))  float  v8f;
typedef __attribute__((ext_vector_type(4)))  unsigned int u32x4;
typedef __attribute__((ext_vector_type(8)))  int          i32x8;
typedef __attribute__((ext_vector_type(4)))  int          i32x4;

constexpr int    Bc = 4, Sc = 2048, Dc = 768, Hc = 12, DFFc = 3072, DKc = 64;
constexpr size_t MT = (size_t)Bc * Sc;   // 8192 token rows

__device__ __forceinline__ bf16 f2bf(float f) {
  unsigned u = __float_as_uint(f);
  unsigned r = u + 0x7fffu + ((u >> 16) & 1u);   // round-to-nearest-even
  unsigned short h = (unsigned short)(r >> 16);
  return __builtin_bit_cast(bf16, h);
}

// ---------------- Tensor Data Mover: 2D bf16 tile -> LDS -------------------
// Builds a D# per CDNA5 ISA ch.8 and issues TENSOR_LOAD_TO_LDS.
// (6-arg clang-23 builtin: g0, g1, g2, g3, extra group, cpol)
// pad_interval_code: pad after 2^(c+1) DWORDs; pad_amount_code: (c+1) DWORDs.
__device__ __forceinline__ void tdm_load_2d(unsigned lds_off, const void* gp,
                                            unsigned tensor_w, unsigned tensor_h,
                                            unsigned row_stride,
                                            unsigned tile_w, unsigned tile_h,
                                            unsigned pad_iv, unsigned pad_am) {
  const unsigned long long ga = (unsigned long long)(uintptr_t)gp;
  u32x4 g0;
  g0[0] = 1u;                                        // count=1, user descriptor
  g0[1] = lds_off;                                   // LDS byte address
  g0[2] = (unsigned)(ga & 0xffffffffu);              // global_addr[31:0]
  g0[3] = (unsigned)((ga >> 32) & 0x01ffffffu)       // global_addr[56:32]
          | 0x80000000u;                             // type=2 ("image")
  i32x8 g1;
  g1[0] = (int)((1u << 16)                           // data_size = 2 bytes
                | (1u << 20)                         // pad_enable
                | (pad_iv << 22) | (pad_am << 25));
  g1[1] = (int)((tensor_w & 0xffffu) << 16);         // dim0[15:0] (abar=0)
  g1[2] = (int)((tensor_w >> 16) | ((tensor_h & 0xffffu) << 16));
  g1[3] = (int)((tensor_h >> 16) | (tile_w << 16));  // dim1 hi | tile_dim0
  g1[4] = (int)(tile_h & 0xffffu);                   // tile_dim1, tile_dim2=0
  g1[5] = (int)row_stride;                           // dim0_stride[31:0]
  g1[6] = 0;                                         // stride hi, dim1_stride
  g1[7] = 0;
  i32x4 gz4;
  gz4[0] = 0; gz4[1] = 0; gz4[2] = 0; gz4[3] = 0;    // groups 2/3: unused dims
  i32x8 gz8;
  for (int e = 0; e < 8; ++e) gz8[e] = 0;            // extra group: zeros
  __builtin_amdgcn_tensor_load_to_lds(g0, g1, gz4, gz4, gz8, 0);
}

// --------------------------- fp32 -> bf16 conversion -----------------------
__global__ void cvt_f32_to_bf16(const float* __restrict__ in,
                                bf16* __restrict__ out, int n) {
  int i = blockIdx.x * blockDim.x + threadIdx.x;
  if (i < n) out[i] = f2bf(in[i]);
}

// ------------------------------ WMMA GEMM ----------------------------------
// C[M,N] = A[M,K] @ W[K,N]   (A,W bf16 row-major; acc fp32)
// MODE 0: fp32 out (+optional bias); 1: bf16 + bias + ReLU; 2: QKV scatter.
// Block: 256 thr = 8 waves; block tile 128x64; wave tile 32x32; K-step 32.
// Tiles stream HBM->LDS via TDM, double buffered (DMA overlaps WMMA).
template <int MODE>
__global__ __launch_bounds__(256)
void gemm_bf16_wmma(const bf16* __restrict__ A, const bf16* __restrict__ W,
                    float* __restrict__ Cf, bf16* __restrict__ Cb,
                    const float* __restrict__ bias, int M, int N, int K) {
  __shared__ __align__(16) bf16 As[2][128][40];  // 128x32 + TDM pad (4 DW/row)
  __shared__ __align__(16) bf16 Bs[2][32][72];   // 32x64  + TDM pad

  const int tid  = threadIdx.x;
  const int lane = tid & 31, wid = tid >> 5;
  const int wm = wid >> 1, wn = wid & 1;         // 4x2 wave grid
  const int ln = lane & 15, lh = lane >> 4;      // wave32 lane split
  const int n0 = blockIdx.x * 64;
  const int m0 = blockIdx.y * 128;

  v8f acc[2][2];
  for (int i = 0; i < 2; ++i)
    for (int j = 0; j < 2; ++j)
      for (int e = 0; e < 8; ++e) acc[i][j][e] = 0.f;

  // Every wave issues identical descriptors (idempotent LDS writes): TDM
  // ignores EXEC, so this is safe under any divergence lowering, and each
  // wave's s_wait_tensorcnt tracks its own copies (L2-resident after first).
  auto tile_issue = [&](int k0, int buf) {
    tdm_load_2d((unsigned)(uintptr_t)&As[buf][0][0],
                A + (size_t)m0 * K + k0,
                (unsigned)K, (unsigned)M, (unsigned)K,
                32u, 128u, 3u /*16 DW*/, 3u /*4 DW*/);
    tdm_load_2d((unsigned)(uintptr_t)&Bs[buf][0][0],
                W + (size_t)k0 * N + n0,
                (unsigned)N, (unsigned)K, (unsigned)N,
                64u, 32u, 4u /*32 DW*/, 3u /*4 DW*/);
  };
  tile_issue(0, 0);

  for (int k0 = 0; k0 < K; k0 += 32) {
    const int cur = (k0 >> 5) & 1;
    __builtin_amdgcn_s_wait_tensorcnt(0);        // current buffer landed
    __syncthreads();
    if (k0 + 32 < K) tile_issue(k0 + 32, cur ^ 1);   // prefetch next via TDM

    // A fragments (ISA 16-bit A 16x32 layout)
    v16bf afr[2], bfr[2];
#pragma unroll
    for (int i = 0; i < 2; ++i) {
      const int r = wm * 32 + i * 16 + ln;
#pragma unroll
      for (int v = 0; v < 8; ++v) {
        const int kk = ((v < 4) ? 0 : 16) + 8 * lh + 2 * (v & 3);
        afr[i][2 * v]     = As[cur][r][kk];
        afr[i][2 * v + 1] = As[cur][r][kk + 1];
      }
    }
    // B fragments (ISA 16-bit B 32x16 layout)
#pragma unroll
    for (int j = 0; j < 2; ++j) {
      const int c = wn * 32 + j * 16 + ln;
#pragma unroll
      for (int v = 0; v < 8; ++v) {
        const int kk = ((v < 4) ? 0 : 16) + 8 * lh + 2 * (v & 3);
        bfr[j][2 * v]     = Bs[cur][kk][c];
        bfr[j][2 * v + 1] = Bs[cur][kk + 1][c];
      }
    }
#pragma unroll
    for (int i = 0; i < 2; ++i)
#pragma unroll
      for (int j = 0; j < 2; ++j)
        acc[i][j] = __builtin_amdgcn_wmma_f32_16x16x32_bf16(
            false, afr[i], false, bfr[j], (short)0, acc[i][j], false, false);
    __syncthreads();                             // cur consumed before reuse
  }

  // fused epilogue  (C layout: VGPR r -> row 8*lh + r, col = lane&15)
#pragma unroll
  for (int i = 0; i < 2; ++i) {
#pragma unroll
    for (int j = 0; j < 2; ++j) {
      const int col = n0 + wn * 32 + j * 16 + ln;
      const float bv = (MODE == 2 || bias == nullptr) ? 0.f : bias[col];
#pragma unroll
      for (int r = 0; r < 8; ++r) {
        const int row = m0 + wm * 32 + i * 16 + lh * 8 + r;
        float val = acc[i][j][r] + bv;
        if (MODE == 0) {
          Cf[(size_t)row * N + col] = val;
        } else if (MODE == 1) {
          val = fmaxf(val, 0.f);
          Cb[(size_t)row * N + col] = f2bf(val);
        } else {  // scatter [M=B*S, N=D] -> [B,H,S,DK]
          const int b = row / Sc, s = row % Sc;
          const int h = col >> 6, dk = col & 63;
          Cb[(((size_t)b * Hc + h) * Sc + s) * DKc + dk] = f2bf(val);
        }
      }
    }
  }
}

// --------------------------- flash attention -------------------------------
// Q,K,V bf16 [B,H,S,DK]; per wave: 16 q-rows, stream 32-key chunks (TDM,
// double buffered). scores=(Q K^T)/8, mask, online softmax, ctx += P V.
__global__ __launch_bounds__(256)
void flash_attn_bf16(const bf16* __restrict__ Q, const bf16* __restrict__ Km,
                     const bf16* __restrict__ Vm, const int* __restrict__ smask,
                     bf16* __restrict__ ctx) {
  __shared__ __align__(16) bf16 Ks[2][32][72];   // 32 keys x 64 d + TDM pad
  __shared__ __align__(16) bf16 Vs[2][32][72];
  __shared__ bf16 Ps[8][16][32];                 // per-wave P staging

  const int tid  = threadIdx.x;
  const int lane = tid & 31, wid = tid >> 5;
  const int ln = lane & 15, lh = lane >> 4;
  const int bh = blockIdx.y;
  const int b = bh / Hc, h = bh % Hc;
  const int q0 = blockIdx.x * 128 + wid * 16;

  const bf16* Qb = Q  + (size_t)bh * Sc * DKc;
  const bf16* Kb = Km + (size_t)bh * Sc * DKc;
  const bf16* Vb = Vm + (size_t)bh * Sc * DKc;

  // Q A-fragments, two K=32 blocks over DK=64 (loaded once)
  v16bf aQ[2];
#pragma unroll
  for (int d = 0; d < 2; ++d)
#pragma unroll
    for (int v = 0; v < 8; ++v) {
      const int kk = ((v < 4) ? 0 : 16) + 8 * lh + 2 * (v & 3);
      const size_t base = (size_t)(q0 + ln) * DKc + d * 32 + kk;
      aQ[d][2 * v]     = Qb[base];
      aQ[d][2 * v + 1] = Qb[base + 1];
    }

  v8f ctxacc[4];
  for (int t = 0; t < 4; ++t)
    for (int e = 0; e < 8; ++e) ctxacc[t][e] = 0.f;
  float mrow[8], lsum[8];
  for (int r = 0; r < 8; ++r) { mrow[r] = -3.0e38f; lsum[r] = 0.f; }

  auto kv_issue = [&](int jj, int buf) {
    tdm_load_2d((unsigned)(uintptr_t)&Ks[buf][0][0], Kb + (size_t)jj * DKc,
                (unsigned)DKc, (unsigned)Sc, (unsigned)DKc,
                64u, 32u, 4u /*32 DW*/, 3u /*4 DW*/);
    tdm_load_2d((unsigned)(uintptr_t)&Vs[buf][0][0], Vb + (size_t)jj * DKc,
                (unsigned)DKc, (unsigned)Sc, (unsigned)DKc,
                64u, 32u, 4u, 3u);
  };
  kv_issue(0, 0);

  for (int j = 0; j < Sc; j += 32) {
    const int cur = (j >> 5) & 1;
    __builtin_amdgcn_s_wait_tensorcnt(0);        // K/V chunk landed
    __syncthreads();
    if (j + 32 < Sc) kv_issue(j + 32, cur ^ 1);  // prefetch next chunk

    // scores: two 16-key tiles, each 2 WMMA over DK
    v8f sc[2];
    for (int jt = 0; jt < 2; ++jt)
      for (int e = 0; e < 8; ++e) sc[jt][e] = 0.f;
#pragma unroll
    for (int jt = 0; jt < 2; ++jt)
#pragma unroll
      for (int d = 0; d < 2; ++d) {
        v16bf bfr;   // B = K^T: rows d, cols key
#pragma unroll
        for (int v = 0; v < 8; ++v) {
          const int dd = d * 32 + ((v < 4) ? 0 : 16) + 8 * lh + 2 * (v & 3);
          bfr[2 * v]     = Ks[cur][jt * 16 + ln][dd];
          bfr[2 * v + 1] = Ks[cur][jt * 16 + ln][dd + 1];
        }
        sc[jt] = __builtin_amdgcn_wmma_f32_16x16x32_bf16(
            false, aQ[d], false, bfr, (short)0, sc[jt], false, false);
      }

    const int mv0 = smask[(size_t)b * Sc + j + ln];
    const int mv1 = smask[(size_t)b * Sc + j + 16 + ln];

    // online softmax: rows live on 16-lane halves -> shfl_xor reductions
#pragma unroll
    for (int r = 0; r < 8; ++r) {
      float s0 = (mv0 == 0) ? -2.0e9f : sc[0][r] * 0.125f;
      float s1 = (mv1 == 0) ? -2.0e9f : sc[1][r] * 0.125f;
      float cm = fmaxf(s0, s1);
#pragma unroll
      for (int mS = 1; mS < 16; mS <<= 1) cm = fmaxf(cm, __shfl_xor(cm, mS, 32));
      const float mnew = fmaxf(mrow[r], cm);
      const float corr = __expf(mrow[r] - mnew);
      const float p0 = __expf(s0 - mnew);
      const float p1 = __expf(s1 - mnew);
      float ps = p0 + p1;
#pragma unroll
      for (int mS = 1; mS < 16; mS <<= 1) ps += __shfl_xor(ps, mS, 32);
      lsum[r] = lsum[r] * corr + ps;
      mrow[r] = mnew;
#pragma unroll
      for (int t = 0; t < 4; ++t) ctxacc[t][r] *= corr;
      Ps[wid][8 * lh + r][ln]      = f2bf(p0);
      Ps[wid][8 * lh + r][16 + ln] = f2bf(p1);
    }
    // wave-private LDS transpose: drain DS counter before A-layout re-read
    asm volatile("s_wait_dscnt 0" ::: "memory");

    v16bf pfr;   // P as A-fragment (16x32)
#pragma unroll
    for (int v = 0; v < 8; ++v) {
      const int kk = ((v < 4) ? 0 : 16) + 8 * lh + 2 * (v & 3);
      pfr[2 * v]     = Ps[wid][ln][kk];
      pfr[2 * v + 1] = Ps[wid][ln][kk + 1];
    }
#pragma unroll
    for (int t = 0; t < 4; ++t) {
      v16bf bfr;   // V 32x16 tile
#pragma unroll
      for (int v = 0; v < 8; ++v) {
        const int kk = ((v < 4) ? 0 : 16) + 8 * lh + 2 * (v & 3);
        bfr[2 * v]     = Vs[cur][kk][t * 16 + ln];
        bfr[2 * v + 1] = Vs[cur][kk + 1][t * 16 + ln];
      }
      ctxacc[t] = __builtin_amdgcn_wmma_f32_16x16x32_bf16(
          false, pfr, false, bfr, (short)0, ctxacc[t], false, false);
    }
    __syncthreads();                             // cur consumed before reuse
  }

  // normalize and write ctx as bf16 [B,S,D]
#pragma unroll
  for (int t = 0; t < 4; ++t)
#pragma unroll
    for (int r = 0; r < 8; ++r) {
      const int s = q0 + lh * 8 + r;
      const float val = ctxacc[t][r] / lsum[r];
      ctx[((size_t)b * Sc + s) * Dc + h * DKc + t * 16 + ln] = f2bf(val);
    }
}

// --------------------- residual add + LayerNorm (ddof=1) -------------------
__global__ __launch_bounds__(256)
void add_ln_kernel(const float* __restrict__ a, const float* __restrict__ bres,
                   const float* __restrict__ alpha, const float* __restrict__ beta,
                   float* __restrict__ outf, bf16* __restrict__ outb) {
  const int row = blockIdx.x, tid = threadIdx.x;
  const float* pa = a + (size_t)row * Dc;
  const float* pb = bres + (size_t)row * Dc;
  __shared__ float red[256];

  float v[3], s = 0.f;
#pragma unroll
  for (int i = 0; i < 3; ++i) {
    const int c = tid + i * 256;
    v[i] = pa[c] + pb[c];
    s += v[i];
  }
  red[tid] = s;
  __syncthreads();
  for (int st = 128; st > 0; st >>= 1) {
    if (tid < st) red[tid] += red[tid + st];
    __syncthreads();
  }
  const float mean = red[0] / (float)Dc;
  __syncthreads();

  float ss = 0.f;
#pragma unroll
  for (int i = 0; i < 3; ++i) {
    const float d = v[i] - mean;
    ss += d * d;
  }
  red[tid] = ss;
  __syncthreads();
  for (int st = 128; st > 0; st >>= 1) {
    if (tid < st) red[tid] += red[tid + st];
    __syncthreads();
  }
  const float stdv = sqrtf(red[0] / (float)(Dc - 1));   // unbiased, torch.std
  const float inv  = 1.f / (stdv + 1e-6f);

#pragma unroll
  for (int i = 0; i < 3; ++i) {
    const int c = tid + i * 256;
    const float o = alpha[c] * (v[i] - mean) * inv + beta[c];
    outf[(size_t)row * Dc + c] = o;
    if (outb) outb[(size_t)row * Dc + c] = f2bf(o);
  }
}

// ------------------------------- launcher ----------------------------------
extern "C" void kernel_launch(void* const* d_in, const int* in_sizes, int n_in,
                              void* d_out, int out_size, void* d_ws, size_t ws_size,
                              hipStream_t stream) {
  (void)in_sizes; (void)n_in; (void)out_size; (void)ws_size;
  const float* x     = (const float*)d_in[0];
  const int*   smask = (const int*)d_in[1];
  const float* wq    = (const float*)d_in[2];
  const float* wk    = (const float*)d_in[3];
  const float* wv    = (const float*)d_in[4];
  const float* wo    = (const float*)d_in[5];
  const float* w1    = (const float*)d_in[6];
  const float* b1    = (const float*)d_in[7];
  const float* w2    = (const float*)d_in[8];
  const float* b2    = (const float*)d_in[9];
  const float* ln1a  = (const float*)d_in[10];
  const float* ln1b  = (const float*)d_in[11];
  const float* ln2a  = (const float*)d_in[12];
  const float* ln2b  = (const float*)d_in[13];
  float* out = (float*)d_out;

  char* wsp = (char*)d_ws;
  size_t off = 0;
  auto take = [&](size_t bytes) -> char* {
    char* p = wsp + off;
    off += (bytes + 255) & ~(size_t)255;
    return p;
  };

  bf16* xb   = (bf16*)take(MT * Dc * 2);
  bf16* wqb  = (bf16*)take((size_t)Dc * Dc * 2);
  bf16* wkb  = (bf16*)take((size_t)Dc * Dc * 2);
  bf16* wvb  = (bf16*)take((size_t)Dc * Dc * 2);
  bf16* wob  = (bf16*)take((size_t)Dc * Dc * 2);
  bf16* w1b  = (bf16*)take((size_t)Dc * DFFc * 2);
  bf16* w2b  = (bf16*)take((size_t)DFFc * Dc * 2);
  bf16* qb   = (bf16*)take(MT * Dc * 2);     // [B,H,S,DK]
  bf16* kb   = (bf16*)take(MT * Dc * 2);
  bf16* vb   = (bf16*)take(MT * Dc * 2);
  bf16* ctxb = (bf16*)take(MT * Dc * 2);     // [B,S,D]
  bf16* x1b  = (bf16*)take(MT * Dc * 2);
  float* attn_out = (float*)take(MT * Dc * 4);
  float* x1       = (float*)take(MT * Dc * 4);
  bf16*  h1b = qb;        // alias: q/k/v/ctx region == exactly MT*DFF*2 bytes
  float* ff  = attn_out;  // alias: attn_out consumed by LN1 before FFN2

  dim3 blk(256);
  auto cvt = [&](const float* src, bf16* dst, size_t n) {
    cvt_f32_to_bf16<<<dim3((unsigned)((n + 255) / 256)), blk, 0, stream>>>(
        src, dst, (int)n);
  };
  cvt(x,  xb,  MT * Dc);
  cvt(wq, wqb, (size_t)Dc * Dc);
  cvt(wk, wkb, (size_t)Dc * Dc);
  cvt(wv, wvb, (size_t)Dc * Dc);
  cvt(wo, wob, (size_t)Dc * Dc);
  cvt(w1, w1b, (size_t)Dc * DFFc);
  cvt(w2, w2b, (size_t)DFFc * Dc);

  const dim3 gD(Dc / 64, (unsigned)(MT / 128));     // N=768 GEMMs
  const dim3 gF(DFFc / 64, (unsigned)(MT / 128));   // N=3072 GEMM

  // QKV projections with fused head-split scatter
  gemm_bf16_wmma<2><<<gD, blk, 0, stream>>>(xb, wqb, nullptr, qb, nullptr,
                                            (int)MT, Dc, Dc);
  gemm_bf16_wmma<2><<<gD, blk, 0, stream>>>(xb, wkb, nullptr, kb, nullptr,
                                            (int)MT, Dc, Dc);
  gemm_bf16_wmma<2><<<gD, blk, 0, stream>>>(xb, wvb, nullptr, vb, nullptr,
                                            (int)MT, Dc, Dc);

  flash_attn_bf16<<<dim3(Sc / 128, Bc * Hc), blk, 0, stream>>>(qb, kb, vb,
                                                               smask, ctxb);

  gemm_bf16_wmma<0><<<gD, blk, 0, stream>>>(ctxb, wob, attn_out, nullptr,
                                            nullptr, (int)MT, Dc, Dc);
  add_ln_kernel<<<dim3((unsigned)MT), blk, 0, stream>>>(x, attn_out, ln1a, ln1b,
                                                        x1, x1b);

  gemm_bf16_wmma<1><<<gF, blk, 0, stream>>>(x1b, w1b, nullptr, h1b, b1,
                                            (int)MT, DFFc, Dc);
  gemm_bf16_wmma<0><<<gD, blk, 0, stream>>>(h1b, w2b, ff, nullptr, b2,
                                            (int)MT, Dc, DFFc);
  add_ln_kernel<<<dim3((unsigned)MT), blk, 0, stream>>>(x1, ff, ln2a, ln2b,
                                                        out, nullptr);
}
